// TransformerASR_20461224198347
// MI455X (gfx1250) — compile-verified, matrix operand order checked by reading, not run on hardware
//
#include <hip/hip_runtime.h>

// ---------------------------------------------------------------------------
// CDNA5 (gfx1250, wave32) Transformer-ASR forward.
// All GEMM-shaped work goes through v_wmma_f32_16x16x32_f16 (f16 in, f32 acc).
// Every GEMM operand is contiguous in K (V is pre-transposed), so fragment
// loads are float4 -> global_load_b128. Each wave computes a 16x(16*JT) tile.
// conv halo handling is branchless (clamped address + cndmask zero).
// ---------------------------------------------------------------------------

typedef __attribute__((ext_vector_type(16))) _Float16 v16h;
typedef __attribute__((ext_vector_type(8)))  float    v8f;

// Contiguous WMMA fragment half-row: elements 0..7 = K +0..7, 8..15 = K +16..23
// (per CDNA5 16-bit 16x32 VGPR layout; caller folds the lane's koff into p).
__device__ __forceinline__ v16h load_frag(const float* __restrict__ p)
{
    const float4* q = (const float4*)p;
    float4 x0 = q[0], x1 = q[1], x2 = q[4], x3 = q[5];
    v16h r;
    r[0]=(_Float16)x0.x; r[1]=(_Float16)x0.y; r[2]=(_Float16)x0.z; r[3]=(_Float16)x0.w;
    r[4]=(_Float16)x1.x; r[5]=(_Float16)x1.y; r[6]=(_Float16)x1.z; r[7]=(_Float16)x1.w;
    r[8]=(_Float16)x2.x; r[9]=(_Float16)x2.y; r[10]=(_Float16)x2.z; r[11]=(_Float16)x2.w;
    r[12]=(_Float16)x3.x; r[13]=(_Float16)x3.y; r[14]=(_Float16)x3.z; r[15]=(_Float16)x3.w;
    return r;
}

// Generic strided batched GEMM: C[z][m][n] = act(scale * sum_k A[m,k]*B[n,k] + bias[n])
// z = (b,h) with b = z/NH; base offsets b*s?b + h*s?h. K contiguous for A and B.
// One wave computes a 16 x (16*JT) tile: 1 A-frag reused across JT B-frags/WMMAs.
template<int JT>
__global__ void __launch_bounds__(256)
gemm_wmma(const float* __restrict__ A, const float* __restrict__ B,
          float* __restrict__ C, const float* __restrict__ bias,
          long sAb, long sAh, long ldam,
          long sBb, long sBh, long ldbn,
          long sCb, long sCh, long ldcm,
          int M, int N, int K, int NH, float scale, int relu)
{
    int z  = blockIdx.z;
    int zb = z / NH, zh = z - zb * NH;
    const float* Az = A + (long)zb * sAb + (long)zh * sAh;
    const float* Bz = B + (long)zb * sBb + (long)zh * sBh;
    float*       Cz = C + (long)zb * sCb + (long)zh * sCh;

    const int GRP = 16 * JT;
    int ntn  = (N + GRP - 1) / GRP;
    int ntm  = M >> 4;
    int wave = blockIdx.x * 8 + (threadIdx.x >> 5);
    if (wave >= ntm * ntn) return;
    int tm = wave / ntn, tn = wave - tm * ntn;

    int lane = threadIdx.x & 31;
    int hs = lane >> 4, l16 = lane & 15, koff = hs * 8;

    const float* Ap = Az + (long)(tm * 16 + l16) * ldam + koff;
    const float* Bp[JT];
#pragma unroll
    for (int j = 0; j < JT; ++j) {
        int br = tn * GRP + j * 16 + l16; if (br > N - 1) br = N - 1;  // pad col
        Bp[j] = Bz + (long)br * ldbn + koff;
    }

    v8f acc[JT]; v8f zero = {};
#pragma unroll
    for (int j = 0; j < JT; ++j) acc[j] = zero;

    for (int k0 = 0; k0 < K; k0 += 32) {
        v16h a = load_frag(Ap + k0);
#pragma unroll
        for (int j = 0; j < JT; ++j) {
            v16h bf = load_frag(Bp[j] + k0);
            acc[j] = __builtin_amdgcn_wmma_f32_16x16x32_f16(false, a, false, bf,
                                                            (short)0, acc[j], false, false);
        }
    }

#pragma unroll
    for (int j = 0; j < JT; ++j) {
        int n = tn * GRP + j * 16 + l16;
        if (n < N) {
            float bv = bias ? bias[n] : 0.0f;
#pragma unroll
            for (int r = 0; r < 8; ++r) {
                int m = tm * 16 + r + hs * 8;
                float v = acc[j][r] * scale + bv;
                if (relu) v = fmaxf(v, 0.f);
                Cz[(long)m * ldcm + n] = v;
            }
        }
    }
}

// conv1: (B,1,1024,80) -> relu -> f16 (B,256,512,40). 9 MACs/output, VALU.
// Branchless halo: clamp coordinates (always in-bounds) + select zero.
__global__ void __launch_bounds__(256)
conv1_kernel(const float* __restrict__ src, const float* __restrict__ w,
             const float* __restrict__ bias, _Float16* __restrict__ out)
{
    long idx = (long)blockIdx.x * 256 + threadIdx.x;      // 16*256*512*40
    if (idx >= 83886080L) return;
    int x = (int)(idx % 40); long t = idx / 40;
    int y = (int)(t % 512); t /= 512;
    int co = (int)(t % 256); int b = (int)(t / 256);
    float acc = bias[co];
    const float* wc = w + co * 9;
    const float* sb = src + (long)b * 1024 * 80;
#pragma unroll
    for (int dy = 0; dy < 3; ++dy) {
#pragma unroll
        for (int dx = 0; dx < 3; ++dx) {
            int yy = 2 * y - 1 + dy, xx = 2 * x - 1 + dx;
            bool ok = (unsigned)yy < 1024u && (unsigned)xx < 80u;
            int yc = min(max(yy, 0), 1023), xc = min(max(xx, 0), 79);
            float v = sb[yc * 80 + xc];
            acc += wc[dy * 3 + dx] * (ok ? v : 0.0f);
        }
    }
    out[idx] = (_Float16)(acc > 0.f ? acc : 0.f);
}

// conv2 implicit GEMM: M=81920 pixels, N=256, K=2304 (72 WMMA k-steps).
// Wave computes 16x64: the im2col A-frag is amortized over 4 weight tiles.
// Branchless halo loads; incremental k -> (ci, k%9) (no divides in the loop).
// Stores into the reference reshape layout xr[(b*16+co>>4)*16+(co&15)][y*20+x].
__global__ void __launch_bounds__(256)
conv2_wmma(const _Float16* __restrict__ h1, const float* __restrict__ w2,
           const float* __restrict__ b2, float* __restrict__ xr)
{
    const int K = 2304;
    int wave = blockIdx.x * 8 + (threadIdx.x >> 5);
    if (wave >= 5120 * 4) return;                 // ntm=5120, ntn=4 (groups of 64)
    int tm = wave >> 2, tn = wave & 3;
    int lane = threadIdx.x & 31;
    int hs = lane >> 4, l16 = lane & 15, koff = hs * 8;

    int m  = tm * 16 + l16;                       // pixel index
    int px = m % 20; int tt = m / 20;
    int py = tt & 255; int pb = tt >> 8;
    const _Float16* hb = h1 + (long)pb * (256L * 512 * 40);

    const float* wp[4];
#pragma unroll
    for (int j = 0; j < 4; ++j)
        wp[j] = w2 + (long)(tn * 64 + j * 16 + l16) * K + koff;

    // incremental k -> (ci = k/9, r9 = k%9) for the 16 fragment elements
    int ci[16], r9[16];
#pragma unroll
    for (int i = 0; i < 16; ++i) {
        int k = koff + ((i & 8) ? i + 8 : i);     // elems 8..15 -> K +16..23
        ci[i] = k / 9; r9[i] = k - ci[i] * 9;
    }

    v8f acc[4]; v8f zero = {};
#pragma unroll
    for (int j = 0; j < 4; ++j) acc[j] = zero;

    for (int k0 = 0; k0 < K; k0 += 32) {
        v16h a;
#pragma unroll
        for (int i = 0; i < 16; ++i) {
            int dy = r9[i] / 3, dx = r9[i] - dy * 3;
            int yy = 2 * py - 1 + dy, xx = 2 * px - 1 + dx;
            bool ok = (unsigned)yy < 512u && (unsigned)xx < 40u;
            int yc = min(max(yy, 0), 511), xc = min(max(xx, 0), 39);
            _Float16 v = hb[((long)ci[i] * 512 + yc) * 40 + xc];   // always in-bounds
            a[i] = ok ? v : (_Float16)0.0f;
            ci[i] += 3; r9[i] += 5;               // advance k by 32 (32 = 3*9 + 5)
            if (r9[i] >= 9) { r9[i] -= 9; ci[i] += 1; }
        }
#pragma unroll
        for (int j = 0; j < 4; ++j) {
            v16h bf = load_frag(wp[j] + k0);
            acc[j] = __builtin_amdgcn_wmma_f32_16x16x32_f16(false, a, false, bf,
                                                            (short)0, acc[j], false, false);
        }
    }

    int px2[8], py2[8], pb2[8];
#pragma unroll
    for (int r = 0; r < 8; ++r) {
        int m2 = tm * 16 + r + hs * 8;
        px2[r] = m2 % 20; int tt2 = m2 / 20;
        py2[r] = tt2 & 255; pb2[r] = tt2 >> 8;
    }
#pragma unroll
    for (int j = 0; j < 4; ++j) {
        int co = tn * 64 + j * 16 + l16;
        float bv = b2[co];
#pragma unroll
        for (int r = 0; r < 8; ++r) {
            float v = acc[j][r] + bv; v = fmaxf(v, 0.f);
            long row = (long)(pb2[r] * 16 + (co >> 4)) * 16 + (co & 15);
            xr[row * 5120 + py2[r] * 20 + px2[r]] = v;
        }
    }
}

// vt[((b*8+h)*32 + d)*Skv + t] = src[t*tstride + b*bstride + h*32 + d]
__global__ void __launch_bounds__(256)
vtrans_kernel(const float* __restrict__ src, float* __restrict__ dst,
              int Skv, long tstride, long bstride)
{
    long idx = (long)blockIdx.x * 256 + threadIdx.x;
    long total = 16L * 8 * 32 * Skv;
    if (idx >= total) return;
    int t = (int)(idx % Skv);
    long rr = idx / Skv;
    int d = (int)(rr & 31);
    int bh = (int)(rr >> 5);
    int h = bh & 7, b = bh >> 3;
    dst[idx] = src[(long)t * tstride + (long)b * bstride + h * 32 + d];
}

// out = LayerNorm(a + r) * g + be, D = 256, one block per row.
__global__ void __launch_bounds__(256)
ln_add_kernel(const float* __restrict__ a, const float* __restrict__ r,
              const float* __restrict__ g, const float* __restrict__ be,
              float* __restrict__ out)
{
    int row = blockIdx.x, d = threadIdx.x;
    __shared__ float red[256];
    float v = a[(long)row * 256 + d] + r[(long)row * 256 + d];
    red[d] = v; __syncthreads();
    for (int s = 128; s > 0; s >>= 1) { if (d < s) red[d] += red[d + s]; __syncthreads(); }
    float mean = red[0] * (1.0f / 256.0f); __syncthreads();
    float c = v - mean;
    red[d] = c * c; __syncthreads();
    for (int s = 128; s > 0; s >>= 1) { if (d < s) red[d] += red[d + s]; __syncthreads(); }
    float var = red[0] * (1.0f / 256.0f);
    out[(long)row * 256 + d] = c * rsqrtf(var + 1e-5f) * g[d] + be[d];
}

// Row softmax, one wave per row. causal!=0 adds +1.0 where t > s
// (the reference adds a triu(ones) mask, i.e. +1.0, NOT -inf).
__global__ void __launch_bounds__(256)
softmax_kernel(float* __restrict__ p, int rows, int ncols, int srow, int causal)
{
    int row = blockIdx.x * 8 + (threadIdx.x >> 5);
    if (row >= rows) return;
    int lane = threadIdx.x & 31;
    float* pr = p + (long)row * ncols;
    int s = row % srow;
    float mx = -3.4e38f;
    for (int j = lane; j < ncols; j += 32) {
        float v = pr[j] + ((causal && j > s) ? 1.0f : 0.0f);
        mx = fmaxf(mx, v);
    }
    for (int o = 16; o; o >>= 1) mx = fmaxf(mx, __shfl_xor(mx, o, 32));
    float sum = 0.f;
    for (int j = lane; j < ncols; j += 32) {
        float v = pr[j] + ((causal && j > s) ? 1.0f : 0.0f);
        v = expf(v - mx);
        pr[j] = v; sum += v;
    }
    for (int o = 16; o; o >>= 1) sum += __shfl_xor(sum, o, 32);
    float inv = 1.0f / sum;
    for (int j = lane; j < ncols; j += 32) pr[j] *= inv;
}

// x[s][b][d] += PE(s,d)
__global__ void __launch_bounds__(256)
posenc_kernel(float* __restrict__ x)
{
    long idx = (long)blockIdx.x * 256 + threadIdx.x;      // 4096*256
    if (idx >= 4096L * 256) return;
    int d = (int)(idx & 255);
    int s = (int)(idx >> 12);                             // row = s*16+b
    float expo  = (float)(d & ~1) * (1.0f / 256.0f);
    float angle = (float)s * expf(-expo * 9.210340371976184f);   // / 10000^expo
    x[idx] += (d & 1) ? cosf(angle) : sinf(angle);
}

// tgt[l][b][:] = embed[tgt_data[b][l]]
__global__ void __launch_bounds__(256)
embed_kernel(const float* __restrict__ emb, const int* __restrict__ tgt,
             float* __restrict__ out)
{
    long idx = (long)blockIdx.x * 256 + threadIdx.x;      // 2048*256
    if (idx >= 2048L * 256) return;
    int d = (int)(idx & 255);
    int lb = (int)(idx >> 8);
    int l = lb >> 4, b = lb & 15;
    int tok = tgt[b * 128 + l];
    out[idx] = emb[(long)tok * 256 + d];
}

// ---------------------------------------------------------------------------
template<int JT>
static inline void launch_gemm(hipStream_t st,
    const float* A, const float* B, float* C, const float* bias,
    long sAb, long sAh, long ldam,
    long sBb, long sBh, long ldbn,
    long sCb, long sCh, long ldcm,
    int M, int N, int K, int batB, int batH, float scale, int relu)
{
    const int GRP = 16 * JT;
    int ntm = M >> 4, ntn = (N + GRP - 1) / GRP;
    int blocks = (ntm * ntn + 7) / 8;
    dim3 g(blocks, 1, batB * batH);
    gemm_wmma<JT><<<g, 256, 0, st>>>(A, B, C, bias,
        sAb, sAh, ldam, sBb, sBh, ldbn, sCb, sCh, ldcm,
        M, N, K, batH, scale, relu);
}

extern "C" void kernel_launch(void* const* d_in, const int* in_sizes, int n_in,
                              void* d_out, int out_size, void* d_ws, size_t ws_size,
                              hipStream_t stream)
{
    const float* src  = (const float*)d_in[0];
    const int*   tgtd = (const int*)  d_in[1];
    const float* c1w  = (const float*)d_in[2];
    const float* c1b  = (const float*)d_in[3];
    const float* c2w  = (const float*)d_in[4];
    const float* c2b  = (const float*)d_in[5];
    const float* redw = (const float*)d_in[6];
    const float* redb = (const float*)d_in[7];
    const float* eqkvw = (const float*)d_in[8];
    const float* eqkvb = (const float*)d_in[9];
    const float* eoutw = (const float*)d_in[10];
    const float* eoutb = (const float*)d_in[11];
    const float* eln1g = (const float*)d_in[12];
    const float* eln1b = (const float*)d_in[13];
    const float* eln2g = (const float*)d_in[14];
    const float* eln2b = (const float*)d_in[15];
    const float* eff1w = (const float*)d_in[16];
    const float* eff1b = (const float*)d_in[17];
    const float* eff2w = (const float*)d_in[18];
    const float* eff2b = (const float*)d_in[19];
    const float* dsaqkvw = (const float*)d_in[20];
    const float* dsaqkvb = (const float*)d_in[21];
    const float* dsaoutw = (const float*)d_in[22];
    const float* dsaoutb = (const float*)d_in[23];
    const float* dcaqkvw = (const float*)d_in[24];
    const float* dcaqkvb = (const float*)d_in[25];
    const float* dcaoutw = (const float*)d_in[26];
    const float* dcaoutb = (const float*)d_in[27];
    const float* dln1g = (const float*)d_in[28];
    const float* dln1b = (const float*)d_in[29];
    const float* dln2g = (const float*)d_in[30];
    const float* dln2b = (const float*)d_in[31];
    const float* dln3g = (const float*)d_in[32];
    const float* dln3b = (const float*)d_in[33];
    const float* dff1w = (const float*)d_in[34];
    const float* dff1b = (const float*)d_in[35];
    const float* dff2w = (const float*)d_in[36];
    const float* dff2b = (const float*)d_in[37];
    const float* emb  = (const float*)d_in[38];
    const float* outw = (const float*)d_in[39];
    const float* outb = (const float*)d_in[40];
    const float* ctcw = (const float*)d_in[41];
    const float* ctcb = (const float*)d_in[42];

    const float ISQ = 0.17677669529663687f;   // 1/sqrt(32)

    // ---- workspace carve-up (all buffers written before read) -------------
    char* ws = (char*)d_ws; size_t o = 0;
    auto take = [&](size_t bytes) { char* p = ws + o; o = (o + bytes + 255) & ~(size_t)255; return p; };
    _Float16* h1   = (_Float16*)take(83886080UL * 2);     // conv1 out f16 (B,256,512,40)
    float* xr      = (float*)take(4096UL * 5120 * 4);     // conv2 out, reshaped layout
    float* bufX    = (float*)take(4096UL * 256 * 4);      // encoder activations / mem
    float* bufQKV  = (float*)take(4096UL * 768 * 4);
    float* bufS    = (float*)take(128UL * 256 * 256 * 4); // attention scores (max size)
    float* bufVT   = (float*)take(16UL * 8 * 32 * 256 * 4); // transposed V
    float* bufA    = (float*)take(4096UL * 256 * 4);
    float* bufH    = (float*)take(4096UL * 256 * 4);
    float* bufT    = (float*)take(4096UL * 256 * 4);
    float* bufF    = (float*)take(4096UL * 2048 * 4);     // FFN hidden
    float* bufTgt  = (float*)take(2048UL * 256 * 4);      // decoder embeddings
    float* bufY    = (float*)take(2048UL * 256 * 4);
    float* bufHd   = (float*)take(2048UL * 256 * 4);
    float* bufH2   = (float*)take(2048UL * 256 * 4);
    float* bufTd   = (float*)take(2048UL * 256 * 4);
    float* bufT2   = (float*)take(2048UL * 256 * 4);
    float* bufQd   = (float*)take(2048UL * 256 * 4);
    float* bufKV   = (float*)take(4096UL * 512 * 4);
    float* bufQKVd = (float*)take(2048UL * 768 * 4);

    float* dec_out = (float*)d_out;                       // (128,16,1001)
    float* ctc_out = (float*)d_out + 128L * 16 * 1001;    // (256,16,1001)

    // ---- front-end --------------------------------------------------------
    conv1_kernel<<<dim3(327680), dim3(256), 0, stream>>>(src, c1w, c1b, h1);
    conv2_wmma <<<dim3(2560),   dim3(256), 0, stream>>>(h1, c2w, c2b, xr);
    // reduction: x = xr @ red_w^T + red_b   (M=4096, N=256, K=5120)
    launch_gemm<4>(stream, xr, redw, bufX, redb,
                   0,0, 5120,  0,0, 5120,  0,0, 256,
                   4096, 256, 5120, 1, 1, 1.0f, 0);
    posenc_kernel<<<dim3(4096), dim3(256), 0, stream>>>(bufX);

    // ---- encoder ----------------------------------------------------------
    for (int i = 0; i < 2; ++i) {
        launch_gemm<4>(stream, bufX, eqkvw + (size_t)i*768*256, bufQKV,
                       eqkvb + (size_t)i*768,
                       0,0, 256,  0,0, 256,  0,0, 768,
                       4096, 768, 256, 1, 1, 1.0f, 0);
        // scores[b,h] = q @ k^T / sqrt(hd)   (M=N=256, K=32)
        launch_gemm<4>(stream, bufQKV, bufQKV + 256, bufS, nullptr,
                       768,32, 12288,  768,32, 12288,  524288,65536, 256,
                       256, 256, 32, 16, 8, ISQ, 0);
        softmax_kernel<<<dim3(4096), dim3(256), 0, stream>>>(bufS, 32768, 256, 256, 0);
        // V^T then o[b,h] = attn @ v   (M=256, N=32, K=256, all contiguous)
        vtrans_kernel<<<dim3(4096), dim3(256), 0, stream>>>(bufQKV + 512, bufVT,
                       256, 12288, 768);
        launch_gemm<2>(stream, bufS, bufVT, bufA, nullptr,
                       524288,65536, 256,  65536,8192, 256,  256,32, 4096,
                       256, 32, 256, 16, 8, 1.0f, 0);
        // out-proj, residual+LN1
        launch_gemm<4>(stream, bufA, eoutw + (size_t)i*256*256, bufT,
                       eoutb + (size_t)i*256,
                       0,0, 256,  0,0, 256,  0,0, 256,
                       4096, 256, 256, 1, 1, 1.0f, 0);
        ln_add_kernel<<<dim3(4096), dim3(256), 0, stream>>>(bufX, bufT,
                       eln1g + (size_t)i*256, eln1b + (size_t)i*256, bufH);
        // FFN
        launch_gemm<4>(stream, bufH, eff1w + (size_t)i*2048*256, bufF,
                       eff1b + (size_t)i*2048,
                       0,0, 256,  0,0, 256,  0,0, 2048,
                       4096, 2048, 256, 1, 1, 1.0f, 1);
        launch_gemm<4>(stream, bufF, eff2w + (size_t)i*256*2048, bufT,
                       eff2b + (size_t)i*256,
                       0,0, 2048,  0,0, 2048,  0,0, 256,
                       4096, 256, 2048, 1, 1, 1.0f, 0);
        ln_add_kernel<<<dim3(4096), dim3(256), 0, stream>>>(bufH, bufT,
                       eln2g + (size_t)i*256, eln2b + (size_t)i*256, bufX);
    }

    // ---- CTC head (from mem = bufX) ---------------------------------------
    launch_gemm<4>(stream, bufX, ctcw, ctc_out, ctcb,
                   0,0, 256,  0,0, 256,  0,0, 1001,
                   4096, 1001, 256, 1, 1, 1.0f, 0);
    softmax_kernel<<<dim3(512), dim3(256), 0, stream>>>(ctc_out, 4096, 1001, 4096, 0);

    // ---- decoder ----------------------------------------------------------
    embed_kernel<<<dim3(2048), dim3(256), 0, stream>>>(emb, tgtd, bufTgt);

    for (int i = 0; i < 2; ++i) {
        // self-attention reads the ORIGINAL embeddings every layer (ref quirk)
        launch_gemm<4>(stream, bufTgt, dsaqkvw + (size_t)i*768*256, bufQKVd,
                       dsaqkvb + (size_t)i*768,
                       0,0, 256,  0,0, 256,  0,0, 768,
                       2048, 768, 256, 1, 1, 1.0f, 0);
        launch_gemm<4>(stream, bufQKVd, bufQKVd + 256, bufS, nullptr,
                       768,32, 12288,  768,32, 12288,  131072,16384, 128,
                       128, 128, 32, 16, 8, ISQ, 0);
        softmax_kernel<<<dim3(2048), dim3(256), 0, stream>>>(bufS, 16384, 128, 128, 1);
        vtrans_kernel<<<dim3(2048), dim3(256), 0, stream>>>(bufQKVd + 512, bufVT,
                       128, 12288, 768);
        launch_gemm<2>(stream, bufS, bufVT, bufTd, nullptr,
                       131072,16384, 128,  32768,4096, 128,  256,32, 4096,
                       128, 32, 128, 16, 8, 1.0f, 0);
        launch_gemm<4>(stream, bufTd, dsaoutw + (size_t)i*256*256, bufT2,
                       dsaoutb + (size_t)i*256,
                       0,0, 256,  0,0, 256,  0,0, 256,
                       2048, 256, 256, 1, 1, 1.0f, 0);
        ln_add_kernel<<<dim3(2048), dim3(256), 0, stream>>>(bufTgt, bufT2,
                       dln1g + (size_t)i*256, dln1b + (size_t)i*256, bufHd);

        // cross-attention: K/V source is mem for i==0, previous decoder y for i==1
        const float* ysrc = (i == 0) ? bufX : bufY;
        int Skv = (i == 0) ? 256 : 128;
        int Mkv = 16 * Skv;
        launch_gemm<4>(stream, bufHd, dcaqkvw + (size_t)i*768*256, bufQd,
                       dcaqkvb + (size_t)i*768,
                       0,0, 256,  0,0, 256,  0,0, 256,
                       2048, 256, 256, 1, 1, 1.0f, 0);
        launch_gemm<4>(stream, ysrc, dcaqkvw + (size_t)i*768*256 + 256*256, bufKV,
                       dcaqkvb + (size_t)i*768 + 256,
                       0,0, 256,  0,0, 256,  0,0, 512,
                       Mkv, 512, 256, 1, 1, 1.0f, 0);
        long sSh = 128L * Skv, sSb = 8L * sSh;
        launch_gemm<4>(stream, bufQd, bufKV, bufS, nullptr,
                       256,32, 4096,  512,32, 8192,  sSb,sSh, Skv,
                       128, Skv, 32, 16, 8, ISQ, 0);
        softmax_kernel<<<dim3(2048), dim3(256), 0, stream>>>(bufS, 16384, Skv, 128, 0);
        vtrans_kernel<<<dim3(16 * Skv), dim3(256), 0, stream>>>(bufKV + 256, bufVT,
                       Skv, 8192, 512);
        launch_gemm<2>(stream, bufS, bufVT, bufTd, nullptr,
                       sSb,sSh, Skv,  256L*Skv, 32L*Skv, Skv,  256,32, 4096,
                       128, 32, Skv, 16, 8, 1.0f, 0);
        launch_gemm<4>(stream, bufTd, dcaoutw + (size_t)i*256*256, bufT2,
                       dcaoutb + (size_t)i*256,
                       0,0, 256,  0,0, 256,  0,0, 256,
                       2048, 256, 256, 1, 1, 1.0f, 0);
        ln_add_kernel<<<dim3(2048), dim3(256), 0, stream>>>(bufHd, bufT2,
                       dln2g + (size_t)i*256, dln2b + (size_t)i*256, bufH2);

        // FFN
        launch_gemm<4>(stream, bufH2, dff1w + (size_t)i*2048*256, bufF,
                       dff1b + (size_t)i*2048,
                       0,0, 256,  0,0, 256,  0,0, 2048,
                       2048, 2048, 256, 1, 1, 1.0f, 1);
        launch_gemm<4>(stream, bufF, dff2w + (size_t)i*256*2048, bufT2,
                       dff2b + (size_t)i*256,
                       0,0, 2048,  0,0, 2048,  0,0, 256,
                       2048, 256, 2048, 1, 1, 1.0f, 0);
        ln_add_kernel<<<dim3(2048), dim3(256), 0, stream>>>(bufH2, bufT2,
                       dln3g + (size_t)i*256, dln3b + (size_t)i*256, bufY);
    }

    // ---- decoder head -----------------------------------------------------
    launch_gemm<4>(stream, bufY, outw, dec_out, outb,
                   0,0, 256,  0,0, 256,  0,0, 1001,
                   2048, 1001, 256, 1, 1, 1.0f, 0);
    softmax_kernel<<<dim3(256), dim3(256), 0, stream>>>(dec_out, 2048, 1001, 2048, 0);
}
